// LSTM_GAT_52604759441723
// MI455X (gfx1250) — matrix operand order repaired and checked by reading, not compiled, not run on hardware
//
#include <hip/hip_runtime.h>
#include <hip/hip_bf16.h>

#define B_    4
#define T_    24
#define N_    1000
#define S_    4000          // B_ * N_
#define H_    128
#define F_    256           // 2*H_
#define G4H   512           // 4*H_
#define E_    1000
#define ETOT  8000          // B_*E_ + S_
#define HEADS_ 4
#define HD_   64

typedef _Float16 half16 __attribute__((ext_vector_type(16)));
typedef _Float16 half8  __attribute__((ext_vector_type(8)));
typedef float    float8 __attribute__((ext_vector_type(8)));

__device__ __forceinline__ float sigf(float x) { return 1.0f / (1.0f + __expf(-x)); }

// ---- WMMA fragment loaders (CDNA5 16x16x32 f16 layouts, wave32) ----

// A: 16x32 (MxK), row-major source. Lane L<16: M=L, K {0..7,16..23}; L>=16: M=L-16, K {8..15,24..31}.
__device__ __forceinline__ half16 load_a16(const _Float16* __restrict__ base, int ld, int lane) {
    int m  = lane & 15;
    int kb = (lane >> 4) << 3;
    const _Float16* p = base + (size_t)m * ld + kb;
    half8 lo = *(const half8*)p;
    half8 hi = *(const half8*)(p + 16);
    return __builtin_shufflevector(lo, hi, 0, 1, 2, 3, 4, 5, 6, 7, 8, 9, 10, 11, 12, 13, 14, 15);
}

// B: pre-packed fragment layout [ktile][ntile][lane][16 halves]; wp points at the fragment.
__device__ __forceinline__ half16 load_bpack(const _Float16* __restrict__ wp, int lane) {
    const _Float16* p = wp + (size_t)lane * 16;
    half8 lo = *(const half8*)p;
    half8 hi = *(const half8*)(p + 8);
    return __builtin_shufflevector(lo, hi, 0, 1, 2, 3, 4, 5, 6, 7, 8, 9, 10, 11, 12, 13, 14, 15);
}

__device__ __forceinline__ void zero8(float8& v) {
#pragma unroll
    for (int j = 0; j < 8; ++j) v[j] = 0.0f;
}

// ---------------- weight pack: f32 -> f16 WMMA B-fragment layout ----------------
// Output layout: [KK/32][NN/16][32 lanes][16 halves], where B[k][n]:
//   transposed=1: B[k][n] = src[n*KK + k]  (lstm weights, src row-major [NN][KK])
//   transposed=0: B[k][n] = src[k*NN + n]  (gat weights,  src row-major [KK][NN])
__global__ void pack_b_kernel(const float* __restrict__ src, _Float16* __restrict__ dst,
                              int KK, int NN, int transposed) {
    int i = blockIdx.x * blockDim.x + threadIdx.x;
    if (i >= KK * NN) return;
    int h    = i & 15;
    int lane = (i >> 4) & 31;
    int frag = i >> 9;                   // fragment index = kt * (NN/16) + nt
    int nt   = frag % (NN >> 4);
    int kt   = frag / (NN >> 4);
    int k    = kt * 32 + ((lane >> 4) << 3) + (h & 7) + ((h >> 3) << 4);
    int n    = nt * 16 + (lane & 15);
    float v  = transposed ? src[(size_t)n * KK + k] : src[(size_t)k * NN + n];
    dst[i] = (_Float16)v;
}

// ---------------- fused LSTM timestep (WMMA) ----------------
// grid: (S_/32, 2 dirs), block 256 (8 waves). Each block: 32 seqs x 512 gate cols.
// Each wave: 2 M-tiles x 4 N-tiles -> every B fragment feeds two WMMAs.
__global__ __launch_bounds__(256)
void lstm_step_kernel(const _Float16* __restrict__ in_seq,  // [T,S,F_] f16, null for layer 0
                      const float*    __restrict__ x,       // [B,T,N] (layer 0)
                      const _Float16* __restrict__ w_ih,    // packed [2][F_/32][32][512] (this layer)
                      const float*    __restrict__ w_ih0,   // [2,G4H] (layer 0)
                      const _Float16* __restrict__ w_hh,    // packed [2][H_/32][32][512] (this layer)
                      const float*    __restrict__ b_ih,    // [2,G4H]
                      const float*    __restrict__ b_hh,    // [2,G4H]
                      _Float16*       __restrict__ out_seq, // [T,S,2H]
                      float*          __restrict__ c_state, // [2,S,H_]
                      int t_step, int first)
{
    __shared__ float gbuf[32][G4H];                  // 64 KB
    const int dir   = blockIdx.y;
    const int t     = dir ? (T_ - 1 - t_step) : t_step;
    const int row0  = blockIdx.x * 32;
    const int tid   = threadIdx.x;
    const int lane  = tid & 31;
    const int wv    = tid >> 5;                      // 0..7
    const int ncol0 = wv * 64;                       // 4 n-tiles per wave
    const int nt0   = ncol0 >> 4;                    // first global n-tile

    float8 acc[2][4];
#pragma unroll
    for (int m = 0; m < 2; ++m)
#pragma unroll
        for (int i = 0; i < 4; ++i) zero8(acc[m][i]);

    // input-to-hidden projection (layers 1,2): g += in[t] @ W_ih^T
    if (in_seq) {
        const _Float16* a0 = in_seq + ((size_t)t * S_ + row0) * F_;
        const _Float16* w0 = w_ih + (size_t)dir * F_ * G4H;   // packed, per-dir
        for (int k0 = 0; k0 < F_; k0 += 32) {
            half16 am0 = load_a16(a0 + k0, F_, lane);
            half16 am1 = load_a16(a0 + (size_t)16 * F_ + k0, F_, lane);
            const _Float16* wk = w0 + ((size_t)(k0 >> 5) * (G4H >> 4) + nt0) * 512;
#pragma unroll
            for (int nt = 0; nt < 4; ++nt) {
                half16 b = load_bpack(wk + (size_t)nt * 512, lane);
                acc[0][nt] = __builtin_amdgcn_wmma_f32_16x16x32_f16(
                    false, am0, false, b, (short)0, acc[0][nt], false, false);
                acc[1][nt] = __builtin_amdgcn_wmma_f32_16x16x32_f16(
                    false, am1, false, b, (short)0, acc[1][nt], false, false);
            }
        }
    }
    // recurrent projection: g += h_prev @ W_hh^T
    if (!first) {
        const int tp = dir ? (t + 1) : (t - 1);
        const _Float16* a0 = out_seq + ((size_t)tp * S_ + row0) * (2 * H_) + dir * H_;
        const _Float16* w0 = w_hh + (size_t)dir * H_ * G4H;   // packed, per-dir
        for (int k0 = 0; k0 < H_; k0 += 32) {
            half16 am0 = load_a16(a0 + k0, 2 * H_, lane);
            half16 am1 = load_a16(a0 + (size_t)16 * (2 * H_) + k0, 2 * H_, lane);
            const _Float16* wk = w0 + ((size_t)(k0 >> 5) * (G4H >> 4) + nt0) * 512;
#pragma unroll
            for (int nt = 0; nt < 4; ++nt) {
                half16 b = load_bpack(wk + (size_t)nt * 512, lane);
                acc[0][nt] = __builtin_amdgcn_wmma_f32_16x16x32_f16(
                    false, am0, false, b, (short)0, acc[0][nt], false, false);
                acc[1][nt] = __builtin_amdgcn_wmma_f32_16x16x32_f16(
                    false, am1, false, b, (short)0, acc[1][nt], false, false);
            }
        }
    }

    // stage gates to LDS (C/D layout: vgpr r -> row r or r+8 by lane half)
    {
        int rbase = (lane >> 4) << 3;
        int c     = lane & 15;
#pragma unroll
        for (int m = 0; m < 2; ++m)
#pragma unroll
            for (int nt = 0; nt < 4; ++nt)
#pragma unroll
                for (int r = 0; r < 8; ++r)
                    gbuf[m * 16 + rbase + r][ncol0 + nt * 16 + c] = acc[m][nt][r];
    }
    __syncthreads();

    // cell update: each thread handles 16 (row, j) pairs
    for (int i = tid; i < 32 * H_; i += 256) {
        int row = i >> 7;
        int j   = i & (H_ - 1);
        int s   = row0 + row;
        float gi = gbuf[row][j];
        float gf = gbuf[row][H_ + j];
        float gg = gbuf[row][2 * H_ + j];
        float go = gbuf[row][3 * H_ + j];
        if (!in_seq) {                                   // layer 0: rank-1 input term
            int bb = s / N_;
            int nn = s - bb * N_;
            float xv = x[((size_t)bb * T_ + t) * N_ + nn];
            const float* w0 = w_ih0 + dir * G4H;
            gi += xv * w0[j];
            gf += xv * w0[H_ + j];
            gg += xv * w0[2 * H_ + j];
            go += xv * w0[3 * H_ + j];
        }
        const float* bi = b_ih + dir * G4H;
        const float* bh = b_hh + dir * G4H;
        gi += bi[j] + bh[j];
        gf += bi[H_ + j] + bh[H_ + j];
        gg += bi[2 * H_ + j] + bh[2 * H_ + j];
        go += bi[3 * H_ + j] + bh[3 * H_ + j];
        float cp = first ? 0.0f : c_state[((size_t)dir * S_ + s) * H_ + j];
        float cn = sigf(gf) * cp + sigf(gi) * tanhf(gg);
        float hn = sigf(go) * tanhf(cn);
        c_state[((size_t)dir * S_ + s) * H_ + j] = cn;
        out_seq[((size_t)t * S_ + s) * (2 * H_) + dir * H_ + j] = (_Float16)hn;
    }
}

// ---------------- mean over time ----------------
__global__ void time_mean_kernel(const _Float16* __restrict__ seq,
                                 float* __restrict__ z, _Float16* __restrict__ z16) {
    int i = blockIdx.x * blockDim.x + threadIdx.x;   // (s,f) flat
    if (i >= S_ * F_) return;
    float acc = 0.0f;
    for (int t = 0; t < T_; ++t) acc += (float)seq[(size_t)t * S_ * F_ + i];
    acc *= (1.0f / T_);
    z[i]   = acc;
    z16[i] = (_Float16)acc;
}

// ---------------- GAT linear (WMMA GEMM [S,256]x[256,256]) ----------------
// grid: S_/32 blocks, 8 waves; each wave: 2 M-tiles x 2 N-tiles.
__global__ __launch_bounds__(256)
void gat_gemm_kernel(const _Float16* __restrict__ z16,  // [S,F_]
                     const _Float16* __restrict__ w16,  // packed [F_/32][F_/16][32][16]
                     float* __restrict__ hh)            // [S,F_]
{
    const int row0 = blockIdx.x * 32;
    const int lane = threadIdx.x & 31;
    const int wv   = threadIdx.x >> 5;
    const int n0   = wv * 32;                            // 2 n-tiles per wave
    const int nt0  = n0 >> 4;
    float8 acc[2][2];
#pragma unroll
    for (int m = 0; m < 2; ++m)
#pragma unroll
        for (int i = 0; i < 2; ++i) zero8(acc[m][i]);
    const _Float16* a0 = z16 + (size_t)row0 * F_;
    for (int k0 = 0; k0 < F_; k0 += 32) {
        half16 am0 = load_a16(a0 + k0, F_, lane);
        half16 am1 = load_a16(a0 + (size_t)16 * F_ + k0, F_, lane);
        const _Float16* wk = w16 + ((size_t)(k0 >> 5) * (F_ >> 4) + nt0) * 512;
#pragma unroll
        for (int nt = 0; nt < 2; ++nt) {
            half16 b = load_bpack(wk + (size_t)nt * 512, lane);
            acc[0][nt] = __builtin_amdgcn_wmma_f32_16x16x32_f16(
                false, am0, false, b, (short)0, acc[0][nt], false, false);
            acc[1][nt] = __builtin_amdgcn_wmma_f32_16x16x32_f16(
                false, am1, false, b, (short)0, acc[1][nt], false, false);
        }
    }
    int rbase = (lane >> 4) << 3;
    int c     = lane & 15;
#pragma unroll
    for (int m = 0; m < 2; ++m)
#pragma unroll
        for (int nt = 0; nt < 2; ++nt)
#pragma unroll
            for (int r = 0; r < 8; ++r)
                hh[(size_t)(row0 + m * 16 + rbase + r) * F_ + n0 + nt * 16 + c] = acc[m][nt][r];
}

// ---------------- attention scores ----------------
__global__ void att_scores_kernel(const float* __restrict__ hh,
                                  const float* __restrict__ a_s, const float* __restrict__ a_d,
                                  float* __restrict__ asrc, float* __restrict__ adst) {
    int i = blockIdx.x * blockDim.x + threadIdx.x;   // (s, head)
    if (i >= S_ * HEADS_) return;
    int s  = i >> 2;
    int hd = i & 3;
    const float* p  = hh + (size_t)s * F_ + hd * HD_;
    const float* ws = a_s + hd * HD_;
    const float* wd = a_d + hd * HD_;
    float vs = 0.0f, vd = 0.0f;
    for (int d = 0; d < HD_; ++d) { float v = p[d]; vs += v * ws[d]; vd += v * wd[d]; }
    asrc[i] = vs;
    adst[i] = vd;
}

// ---------------- init m/den/agg ----------------
__global__ void gat_init_kernel(float* __restrict__ m, float* __restrict__ den,
                                float* __restrict__ agg) {
    int i = blockIdx.x * blockDim.x + threadIdx.x;
    if (i < S_ * HEADS_) { m[i] = -1e30f; den[i] = 0.0f; }
    if (i < S_ * F_) agg[i] = 0.0f;
}

__device__ __forceinline__ void edge_sd(const int* __restrict__ ei, int g, int& src, int& dst) {
    if (g < B_ * E_) {
        int bb = g / E_;
        int e0 = g - bb * E_;
        src = ei[e0]      + bb * N_;
        dst = ei[E_ + e0] + bb * N_;
    } else {
        src = dst = g - B_ * E_;                     // self loop
    }
}

__device__ __forceinline__ void atomicMaxF(float* addr, float val) {
    int* ai  = (int*)addr;
    int  cur = __float_as_int(*addr);
    while (__int_as_float(cur) < val) {
        int prev = atomicCAS(ai, cur, __float_as_int(val));
        if (prev == cur) break;
        cur = prev;
    }
}

// ---------------- edge softmax: max, exp-sum, aggregate ----------------
__global__ void edge_max_kernel(const int* __restrict__ ei,
                                const float* __restrict__ asrc, const float* __restrict__ adst,
                                float* __restrict__ m) {
    int i = blockIdx.x * blockDim.x + threadIdx.x;   // (edge, head)
    if (i >= ETOT * HEADS_) return;
    int g = i >> 2, h = i & 3, src, dst;
    edge_sd(ei, g, src, dst);
    float ev = asrc[src * HEADS_ + h] + adst[dst * HEADS_ + h];
    ev = ev > 0.0f ? ev : 0.2f * ev;                 // leaky_relu(0.2)
    atomicMaxF(&m[dst * HEADS_ + h], ev);
}

__global__ void edge_expsum_kernel(const int* __restrict__ ei,
                                   const float* __restrict__ asrc, const float* __restrict__ adst,
                                   const float* __restrict__ m,
                                   float* __restrict__ den, float* __restrict__ ee) {
    int i = blockIdx.x * blockDim.x + threadIdx.x;
    if (i >= ETOT * HEADS_) return;
    int g = i >> 2, h = i & 3, src, dst;
    edge_sd(ei, g, src, dst);
    float ev = asrc[src * HEADS_ + h] + adst[dst * HEADS_ + h];
    ev = ev > 0.0f ? ev : 0.2f * ev;
    float e = __expf(ev - m[dst * HEADS_ + h]);
    ee[i] = e;
    atomicAdd(&den[dst * HEADS_ + h], e);
}

__global__ void edge_agg_kernel(const int* __restrict__ ei,
                                const float* __restrict__ hh, const float* __restrict__ ee,
                                const float* __restrict__ den, float* __restrict__ agg) {
    int i = blockIdx.x * blockDim.x + threadIdx.x;   // (edge, feature)
    if (i >= ETOT * F_) return;
    int g = i >> 8, f = i & (F_ - 1), src, dst;
    int h = f >> 6;
    edge_sd(ei, g, src, dst);
    float alpha = ee[g * HEADS_ + h] / (den[dst * HEADS_ + h] + 1e-16f);
    atomicAdd(&agg[(size_t)dst * F_ + f], hh[(size_t)src * F_ + f] * alpha);
}

// ---------------- GraphNorm stats: one wave per (batch, feature) ----------------
__global__ __launch_bounds__(256)
void norm_stats_kernel(const float* __restrict__ agg, const float* __restrict__ bias,
                       float* __restrict__ meanb, float* __restrict__ msqb) {
    int wid  = (blockIdx.x * blockDim.x + threadIdx.x) >> 5;   // 0..B_*F_-1
    int lane = threadIdx.x & 31;
    int bb   = wid >> 8;
    int f    = wid & (F_ - 1);
    float bv = bias[f];
    float sum = 0.0f, sq = 0.0f;
    for (int n = lane; n < N_; n += 32) {
        float v = agg[(size_t)(bb * N_ + n) * F_ + f] + bv;
        sum += v; sq += v * v;
    }
#pragma unroll
    for (int off = 16; off; off >>= 1) {
        sum += __shfl_down(sum, off, 32);
        sq  += __shfl_down(sq,  off, 32);
    }
    if (lane == 0) {
        meanb[bb * F_ + f] = sum * (1.0f / N_);
        msqb[bb * F_ + f]  = sq  * (1.0f / N_);
    }
}

// ---------------- GraphNorm apply + residual relu ----------------
__global__ void norm_apply_kernel(const float* __restrict__ agg, const float* __restrict__ bias,
                                  const float* __restrict__ meanb, const float* __restrict__ msqb,
                                  const float* __restrict__ nw, const float* __restrict__ nb,
                                  const float* __restrict__ nms,
                                  float* __restrict__ z, _Float16* __restrict__ z16) {
    int i = blockIdx.x * blockDim.x + threadIdx.x;   // (s, f)
    if (i >= S_ * F_) return;
    int s  = i >> 8;
    int f  = i & (F_ - 1);
    int bb = s / N_;
    float mean = meanb[bb * F_ + f];
    float msq  = msqb[bb * F_ + f];
    float ms   = nms[f];
    float xv   = agg[i] + bias[f];
    float sub  = xv - ms * mean;
    float var  = msq - mean * mean * (2.0f * ms - ms * ms);   // E[sub^2]
    float y    = nw[f] * sub * rsqrtf(var + 1e-5f) + nb[f];
    float outv = y + z[i];
    outv = outv > 0.0f ? outv : 0.0f;
    z[i]   = outv;
    z16[i] = (_Float16)outv;
}

// ---------------- pooling + classifier ----------------
__global__ __launch_bounds__(256)
void pool_clf_kernel(const float* __restrict__ z, const float* __restrict__ clf_w,
                     const float* __restrict__ clf_b, float* __restrict__ out) {
    __shared__ float red[256];
    int bb = blockIdx.x;
    int f  = threadIdx.x;
    float sum = 0.0f;
    for (int n = 0; n < N_; ++n) sum += z[(size_t)(bb * N_ + n) * F_ + f];
    red[f] = sum * (1.0f / N_) * clf_w[f];
    __syncthreads();
    for (int off = 128; off; off >>= 1) {
        if (f < off) red[f] += red[f + off];
        __syncthreads();
    }
    if (f == 0) out[bb] = red[0] + clf_b[0];
}

extern "C" void kernel_launch(void* const* d_in, const int* in_sizes, int n_in,
                              void* d_out, int out_size, void* d_ws, size_t ws_size,
                              hipStream_t stream) {
    (void)in_sizes; (void)n_in; (void)out_size; (void)ws_size;
    const float* x         = (const float*)d_in[0];
    const int*   ei        = (const int*)d_in[1];
    const float* w_ih0     = (const float*)d_in[2];   // [2,512,1]
    const float* w_ih12    = (const float*)d_in[3];   // [2,2,512,256]
    const float* w_hh      = (const float*)d_in[4];   // [3,2,512,128]
    const float* b_ih      = (const float*)d_in[5];   // [3,2,512]
    const float* b_hh      = (const float*)d_in[6];
    const float* gat_lin_w = (const float*)d_in[7];   // [3,256,256]
    const float* a_src     = (const float*)d_in[8];   // [3,4,64]
    const float* a_dst     = (const float*)d_in[9];
    const float* gat_bias  = (const float*)d_in[10];  // [3,256]
    const float* norm_w    = (const float*)d_in[11];
    const float* norm_b    = (const float*)d_in[12];
    const float* norm_ms   = (const float*)d_in[13];
    const float* clf_w     = (const float*)d_in[14];
    const float* clf_b     = (const float*)d_in[15];
    float* out = (float*)d_out;

    char* p = (char*)d_ws;
    auto alloc = [&](size_t bytes) { char* r = p; p += (bytes + 255) & ~(size_t)255; return r; };
    _Float16* hbufA    = (_Float16*)alloc((size_t)T_ * S_ * F_ * 2);
    _Float16* hbufB    = (_Float16*)alloc((size_t)T_ * S_ * F_ * 2);
    float*    c_state  = (float*)   alloc((size_t)2 * S_ * H_ * 4);
    _Float16* w_ih12_p = (_Float16*)alloc((size_t)2 * 2 * G4H * F_ * 2);  // packed
    _Float16* w_hh_p   = (_Float16*)alloc((size_t)3 * 2 * G4H * H_ * 2);  // packed
    _Float16* gatw_p   = (_Float16*)alloc((size_t)3 * F_ * F_ * 2);       // packed
    float*    z        = (float*)   alloc((size_t)S_ * F_ * 4);
    _Float16* z16      = (_Float16*)alloc((size_t)S_ * F_ * 2);
    float*    hh       = (float*)   alloc((size_t)S_ * F_ * 4);
    float*    agg      = (float*)   alloc((size_t)S_ * F_ * 4);
    float*    asrc     = (float*)   alloc((size_t)S_ * HEADS_ * 4);
    float*    adst     = (float*)   alloc((size_t)S_ * HEADS_ * 4);
    float*    mbuf     = (float*)   alloc((size_t)S_ * HEADS_ * 4);
    float*    den      = (float*)   alloc((size_t)S_ * HEADS_ * 4);
    float*    eebuf    = (float*)   alloc((size_t)ETOT * HEADS_ * 4);
    float*    meanb    = (float*)   alloc((size_t)B_ * F_ * 4);
    float*    msqb     = (float*)   alloc((size_t)B_ * F_ * 4);

    // pack weights into WMMA B-fragment layout (f16)
    {
        // w_ih12: 4 slices of [512 x 256], transposed (B = W^T)
        for (int sidx = 0; sidx < 4; ++sidx) {
            int n = G4H * F_;
            pack_b_kernel<<<(n + 255) / 256, 256, 0, stream>>>(
                w_ih12 + (size_t)sidx * n, w_ih12_p + (size_t)sidx * n, F_, G4H, 1);
        }
        // w_hh: 6 slices of [512 x 128], transposed
        for (int sidx = 0; sidx < 6; ++sidx) {
            int n = G4H * H_;
            pack_b_kernel<<<(n + 255) / 256, 256, 0, stream>>>(
                w_hh + (size_t)sidx * n, w_hh_p + (size_t)sidx * n, H_, G4H, 1);
        }
        // gat_lin_w: 3 slices of [256 x 256], not transposed (B = W)
        for (int sidx = 0; sidx < 3; ++sidx) {
            int n = F_ * F_;
            pack_b_kernel<<<(n + 255) / 256, 256, 0, stream>>>(
                gat_lin_w + (size_t)sidx * n, gatw_p + (size_t)sidx * n, F_, F_, 0);
        }
    }

    // 3-layer bidirectional LSTM (buffers ping-pong A -> B -> A)
    for (int l = 0; l < 3; ++l) {
        const _Float16* in_seq  = (l == 0) ? nullptr : (l == 1 ? hbufA : hbufB);
        _Float16*       out_seq = (l == 1) ? hbufB : hbufA;
        const _Float16* wih = (l == 0) ? nullptr : (w_ih12_p + (size_t)(l - 1) * 2 * G4H * F_);
        const _Float16* whh = w_hh_p + (size_t)l * 2 * G4H * H_;
        const float*    bi  = b_ih + (size_t)l * 2 * G4H;
        const float*    bh  = b_hh + (size_t)l * 2 * G4H;
        for (int ts = 0; ts < T_; ++ts) {
            lstm_step_kernel<<<dim3(S_ / 32, 2), 256, 0, stream>>>(
                in_seq, x, wih, w_ih0, whh, bi, bh, out_seq, c_state, ts, ts == 0 ? 1 : 0);
        }
    }

    // mean over time -> node features (layer-2 output is in hbufA)
    time_mean_kernel<<<(S_ * F_ + 255) / 256, 256, 0, stream>>>(hbufA, z, z16);

    // 3 GAT + GraphNorm + residual ReLU layers
    for (int l = 0; l < 3; ++l) {
        gat_init_kernel<<<(S_ * F_ + 255) / 256, 256, 0, stream>>>(mbuf, den, agg);
        gat_gemm_kernel<<<S_ / 32, 256, 0, stream>>>(z16, gatw_p + (size_t)l * F_ * F_, hh);
        att_scores_kernel<<<(S_ * HEADS_ + 255) / 256, 256, 0, stream>>>(
            hh, a_src + l * HEADS_ * HD_, a_dst + l * HEADS_ * HD_, asrc, adst);
        edge_max_kernel<<<(ETOT * HEADS_ + 255) / 256, 256, 0, stream>>>(ei, asrc, adst, mbuf);
        edge_expsum_kernel<<<(ETOT * HEADS_ + 255) / 256, 256, 0, stream>>>(
            ei, asrc, adst, mbuf, den, eebuf);
        edge_agg_kernel<<<(ETOT * F_) / 256, 256, 0, stream>>>(ei, hh, eebuf, den, agg);
        norm_stats_kernel<<<(B_ * F_ * 32) / 256, 256, 0, stream>>>(
            agg, gat_bias + l * F_, meanb, msqb);
        norm_apply_kernel<<<(S_ * F_ + 255) / 256, 256, 0, stream>>>(
            agg, gat_bias + l * F_, meanb, msqb,
            norm_w + l * F_, norm_b + l * F_, norm_ms + l * F_, z, z16);
    }

    pool_clf_kernel<<<B_, 256, 0, stream>>>(z, clf_w, clf_b, out);
}
